// CRF_11656541241559
// MI455X (gfx1250) — matrix operand order repaired and checked by reading, not compiled, not run on hardware
//
#include <hip/hip_runtime.h>
#include <hip/hip_bf16.h>

#define Bb 512
#define Ss 1024
#define Tt 64

#define LOG2E_F 1.4426950408889634f
#define LN2_F   0.6931471805599453f

typedef _Float16 v16h __attribute__((ext_vector_type(16)));
typedef __fp16   h2   __attribute__((ext_vector_type(2)));
typedef float    v8f  __attribute__((ext_vector_type(8)));

union BFrag  { v16h h; unsigned int u[8]; };
union PackCV { h2 h; unsigned int u; };

// Cross-half (lane ^ 16) exchange as a pure VALU op: v_permlanex16_b32 with
// identity per-lane nibble selectors. Avoids ds_bpermute round-trips.
__device__ __forceinline__ unsigned int xor16_u32(unsigned int v) {
    return (unsigned int)__builtin_amdgcn_permlanex16(
        (int)v, (int)v, 0x76543210, (int)0xFEDCBA98, true, false);
}
__device__ __forceinline__ float xor16_f32(float v) {
    union { float f; int i; } a, r;
    a.f = v;
    r.i = __builtin_amdgcn_permlanex16(a.i, a.i, 0x76543210, (int)0xFEDCBA98,
                                       true, false);
    return r.f;
}

// ---------------------------------------------------------------------------
// Forward (denominator) kernel, transposed formulation:
//   D[j,m] = sum_i A[j,i] * B[i,m],  A = exp(trans)^T (constant, registers),
//   B = p = 2^(alpha2 - colmax) (f16, v_cvt_pk_rtz packed), alpha2 =
// alpha*log2(e) lives in the WMMA D layout (states striped over VGPRs,
// batch over lanes); the next step's B fragment is one v_permlanex16
// half-swap + pack: zero LDS traffic inside the 1023-step serial loop.
// One wave (wave32) per 16 batch rows; 32 blocks x 32 threads.
// ---------------------------------------------------------------------------
__global__ __launch_bounds__(32) void crf_fwd_kernel(
    const float* __restrict__ em, const int* __restrict__ mask,
    const float* __restrict__ startT, const float* __restrict__ endT,
    const float* __restrict__ trans, float* __restrict__ den)
{
    __shared__ __align__(16) _Float16 Elds[Tt * Tt];   // exp(trans)[i][j]

    const int lane = threadIdx.x & 31;
    const int hL   = lane >> 4;       // half: 0 = lanes 0-15, 1 = lanes 16-31
    const int n16  = lane & 15;       // batch column within tile
    const int b    = blockIdx.x * 16 + n16;

    for (int i = threadIdx.x; i < Tt * Tt; i += 32)
        Elds[i] = (_Float16)__expf(trans[i]);
    __syncthreads();

    // --- Constant A fragments: A[j,i] = exp(trans[i][j]); Af[mt][kh] covers
    //     j = mt*16..+15 (M), i = kh*32..+31 (K)
    v16h Af[4][2];
    #pragma unroll
    for (int mt = 0; mt < 4; ++mt) {
        #pragma unroll
        for (int kh = 0; kh < 2; ++kh) {
            v16h f;
            #pragma unroll
            for (int e = 0; e < 16; ++e) {
                int i = kh * 32 + 16 * hL + e;
                int j = mt * 16 + n16;
                f[e] = Elds[i * Tt + j];
            }
            Af[mt][kh] = f;
        }
    }

    // --- alpha2_0[j] = (start[j] + em[0,b,j]) * log2(e), D-layout:
    //     acc[mt][r] at this lane = state j = mt*16 + r + 8*hL, batch b
    v8f acc[4];
    #pragma unroll
    for (int mt = 0; mt < 4; ++mt) {
        const float* e0 = em + (size_t)b * Ss * Tt + mt * 16 + 8 * hL;
        float4 lo = *(const float4*)e0;
        float4 hi = *(const float4*)(e0 + 4);
        float ev[8] = {lo.x, lo.y, lo.z, lo.w, hi.x, hi.y, hi.z, hi.w};
        #pragma unroll
        for (int r = 0; r < 8; ++r) {
            int j = mt * 16 + 8 * hL + r;
            acc[mt][r] = (startT[j] + ev[r]) * LOG2E_F;
        }
    }

    // Running pointers: one u64 add each per iteration instead of rebuilds
    const float* ept = em + ((size_t)b * Ss + 1) * Tt + 8 * hL;
    const float* pft = em + ((size_t)b * Ss + 17) * Tt + hL * 32;  // t+16 ahead
    const int*   mkp = mask + (size_t)b * Ss + 1;

    for (int t = 1; t < Ss; ++t) {
        // ---- issue memory early: mask (1 dword) + em tile (8 x b128) + prefetch
        int mk = *mkp;
        mkp += 1;
        float4 elo[4], ehi[4];
        #pragma unroll
        for (int mt = 0; mt < 4; ++mt) {
            elo[mt] = *(const float4*)(ept + mt * 16);
            ehi[mt] = *(const float4*)(ept + mt * 16 + 4);
        }
        ept += Tt;
        // speculative prefetch (safe past end of buffer: dropped on xlat fail)
        __builtin_prefetch(pft, 0, 1);
        __builtin_prefetch(pft + 16, 0, 1);
        pft += Tt;

        // ---- column max over all 64 states for this batch element
        float tm[4];
        #pragma unroll
        for (int mt = 0; mt < 4; ++mt) {
            v8f a = acc[mt];
            tm[mt] = fmaxf(fmaxf(fmaxf(a[0], a[1]), fmaxf(a[2], a[3])),
                           fmaxf(fmaxf(a[4], a[5]), fmaxf(a[6], a[7])));
        }
        float mx = fmaxf(fmaxf(tm[0], tm[1]), fmaxf(tm[2], tm[3]));
        mx = fmaxf(mx, xor16_f32(mx));

        // ---- emx[mt][r] = em*log2e + mx : independent work, overlaps WMMA
        float emx[4][8];
        #pragma unroll
        for (int mt = 0; mt < 4; ++mt) {
            float ev[8] = {elo[mt].x, elo[mt].y, elo[mt].z, elo[mt].w,
                           ehi[mt].x, ehi[mt].y, ehi[mt].z, ehi[mt].w};
            #pragma unroll
            for (int r = 0; r < 8; ++r)
                emx[mt][r] = fmaf(ev[r], LOG2E_F, mx);
        }

        // ---- p = 2^(alpha2 - mx) -> packed f16 pairs (v_cvt_pk_rtz_f16_f32)
        unsigned int pk[4][4], pp[4][4];
        #pragma unroll
        for (int mt = 0; mt < 4; ++mt) {
            #pragma unroll
            for (int q = 0; q < 4; ++q) {
                float p0 = __builtin_amdgcn_exp2f(acc[mt][2 * q]     - mx);
                float p1 = __builtin_amdgcn_exp2f(acc[mt][2 * q + 1] - mx);
                PackCV cv;
                cv.h = __builtin_amdgcn_cvt_pkrtz(p0, p1);
                pk[mt][q] = cv.u;
            }
        }
        #pragma unroll
        for (int mt = 0; mt < 4; ++mt)
            #pragma unroll
            for (int q = 0; q < 4; ++q)
                pp[mt][q] = xor16_u32(pk[mt][q]);

        // ---- assemble B fragments (i = K rows, batch = N cols)
        //      B[kh] lane-half hL: e0-7 from tile 2kh+hL lanes0-15, e8-15 lanes16-31
        BFrag bf0, bf1;
        #pragma unroll
        for (int q = 0; q < 4; ++q) {
            bf0.u[q]     = hL ? pp[1][q] : pk[0][q];
            bf0.u[4 + q] = hL ? pk[1][q] : pp[0][q];
            bf1.u[q]     = hL ? pp[3][q] : pk[2][q];
            bf1.u[4 + q] = hL ? pk[3][q] : pp[2][q];
        }

        // ---- S = E^T @ p : 8 WMMAs (4 state tiles x K=64 via 2 chained K=32)
        v8f nacc[4];
        #pragma unroll
        for (int mt = 0; mt < 4; ++mt) {
            v8f c = {};
            c = __builtin_amdgcn_wmma_f32_16x16x32_f16(false, Af[mt][0], false, bf0.h,
                                                       (short)0, c, false, false);
            c = __builtin_amdgcn_wmma_f32_16x16x32_f16(false, Af[mt][1], false, bf1.h,
                                                       (short)0, c, false, false);
            nacc[mt] = c;
        }

        // ---- alpha2' = log2(S) + (mx + em*log2e); branchless FMA freeze when
        //      mask==0 (ternary would get if-converted into exec-mask branches)
        float mkf = (float)mk;
        #pragma unroll
        for (int mt = 0; mt < 4; ++mt) {
            #pragma unroll
            for (int r = 0; r < 8; ++r) {
                float an = __builtin_amdgcn_logf(nacc[mt][r]) + emx[mt][r];
                acc[mt][r] = fmaf(mkf, an - acc[mt][r], acc[mt][r]);
            }
        }
    }

    // ---- denominator[b] = ln2 * log2( sum_j 2^(alpha2[j] + end[j]*log2e) )
    #pragma unroll
    for (int mt = 0; mt < 4; ++mt)
        #pragma unroll
        for (int r = 0; r < 8; ++r)
            acc[mt][r] += endT[mt * 16 + 8 * hL + r] * LOG2E_F;

    float tm[4];
    #pragma unroll
    for (int mt = 0; mt < 4; ++mt) {
        v8f a = acc[mt];
        tm[mt] = fmaxf(fmaxf(fmaxf(a[0], a[1]), fmaxf(a[2], a[3])),
                       fmaxf(fmaxf(a[4], a[5]), fmaxf(a[6], a[7])));
    }
    float mx = fmaxf(fmaxf(tm[0], tm[1]), fmaxf(tm[2], tm[3]));
    mx = fmaxf(mx, xor16_f32(mx));

    float s = 0.f;
    #pragma unroll
    for (int mt = 0; mt < 4; ++mt)
        #pragma unroll
        for (int r = 0; r < 8; ++r)
            s += __builtin_amdgcn_exp2f(acc[mt][r] - mx);
    s += xor16_f32(s);

    float d = (__builtin_amdgcn_logf(s) + mx) * LN2_F;
    if (hL == 0) den[b] = d;
}

// ---------------------------------------------------------------------------
// Numerator (gold-path score) kernel: one block per batch element.
// ---------------------------------------------------------------------------
__global__ __launch_bounds__(128) void crf_num_kernel(
    const float* __restrict__ em, const int* __restrict__ tags,
    const int* __restrict__ mask, const float* __restrict__ startT,
    const float* __restrict__ endT, const float* __restrict__ trans,
    float* __restrict__ numv, float* __restrict__ cntv)
{
    const int b   = blockIdx.x;
    const int tid = threadIdx.x;
    float s = 0.f;
    int   c = 0;
    for (int t = tid; t < Ss; t += 128) {
        int mk = mask[(size_t)b * Ss + t];
        c += mk;
        if (t > 0 && mk) {
            int tp = tags[(size_t)b * Ss + t - 1];
            int tc = tags[(size_t)b * Ss + t];
            s += trans[tp * Tt + tc] + em[((size_t)b * Ss + t) * Tt + tc];
        }
    }
    __shared__ float sf[128];
    __shared__ int   si[128];
    sf[tid] = s; si[tid] = c;
    __syncthreads();
    for (int off = 64; off > 0; off >>= 1) {
        if (tid < off) { sf[tid] += sf[tid + off]; si[tid] += si[tid + off]; }
        __syncthreads();
    }
    if (tid == 0) {
        int   cnt    = si[0];
        int   t0     = tags[(size_t)b * Ss];
        float score0 = startT[t0] + em[(size_t)b * Ss * Tt + t0];
        int   last   = tags[(size_t)b * Ss + cnt - 1];
        numv[b] = score0 + sf[0] + endT[last];
        cntv[b] = (float)cnt;
    }
}

// ---------------------------------------------------------------------------
// Final reduction: loss = -sum(num - den) / sum(mask)
// ---------------------------------------------------------------------------
__global__ __launch_bounds__(512) void crf_finish_kernel(
    const float* __restrict__ numv, const float* __restrict__ den,
    const float* __restrict__ cntv, float* __restrict__ out)
{
    __shared__ float sa[512];
    __shared__ float sc[512];
    const int tid = threadIdx.x;
    sa[tid] = numv[tid] - den[tid];
    sc[tid] = cntv[tid];
    __syncthreads();
    for (int off = 256; off > 0; off >>= 1) {
        if (tid < off) { sa[tid] += sa[tid + off]; sc[tid] += sc[tid + off]; }
        __syncthreads();
    }
    if (tid == 0) out[0] = -(sa[0] / sc[0]);
}

extern "C" void kernel_launch(void* const* d_in, const int* in_sizes, int n_in,
                              void* d_out, int out_size, void* d_ws, size_t ws_size,
                              hipStream_t stream)
{
    const float* em     = (const float*)d_in[0];
    const int*   tags   = (const int*)  d_in[1];
    const int*   mask   = (const int*)  d_in[2];
    const float* startT = (const float*)d_in[3];
    const float* endT   = (const float*)d_in[4];
    const float* trans  = (const float*)d_in[5];
    float* out = (float*)d_out;

    float* den  = (float*)d_ws;       // [512]
    float* numv = den + Bb;           // [512]
    float* cntv = numv + Bb;          // [512]

    // 32 waves, one per 16 batch rows, each on its own WGP
    crf_fwd_kernel<<<32, 32, 0, stream>>>(em, mask, startT, endT, trans, den);
    crf_num_kernel<<<Bb, 128, 0, stream>>>(em, tags, mask, startT, endT, trans, numv, cntv);
    crf_finish_kernel<<<1, 512, 0, stream>>>(numv, den, cntv, out);
}
